// TRLayer_53970559041674
// MI455X (gfx1250) — matrix operand order repaired and checked by reading, not compiled, not run on hardware
//
#include <hip/hip_runtime.h>
#include <hip/hip_bf16.h>

typedef __attribute__((ext_vector_type(16))) _Float16 v16h;
typedef __attribute__((ext_vector_type(8)))  _Float16 h8;
typedef __attribute__((ext_vector_type(8)))  float    v8f;
typedef __attribute__((ext_vector_type(4)))  float    f4;

#define RANK 8
#define N0 64
#define N1 64
#define N2 64
#define N3 16
#define KDIM 2048            // x.shape[1] == W rows
#define NDIM 2048            // W cols
#define MDIM 8192            // x rows
#define KTILE 64             // K staging depth per barrier

// ---------------------------------------------------------------------------
// Stage 1a: PA[(i0*64+i1)][a*8+b] = sum_c f0[a,i0,c] * f1[c,i1,b]
// ---------------------------------------------------------------------------
__global__ void tr_pairA(const float* __restrict__ f0, const float* __restrict__ f1,
                         float* __restrict__ PA) {
  int g = blockIdx.x * blockDim.x + threadIdx.x;     // 4096*64 threads
  int pair = g >> 6;          // i0*64 + i1
  int e    = g & 63;          // a*8 + b
  int i0 = pair >> 6, i1 = pair & 63;
  int a = e >> 3, b = e & 7;
  float s = 0.f;
#pragma unroll
  for (int c = 0; c < RANK; ++c)
    s += f0[a * (N0 * RANK) + i0 * RANK + c] * f1[c * (N1 * RANK) + i1 * RANK + b];
  PA[(size_t)pair * 64 + e] = s;
}

// ---------------------------------------------------------------------------
// Stage 1b: PB[(i2*16+i3)][a*8+b] = B[b][a] = sum_c f2[b,i2,c] * f3[c,i3,a]
// (stored pre-transposed so W = dot(PA_row, PB_row) over 64 elements)
// ---------------------------------------------------------------------------
__global__ void tr_pairB(const float* __restrict__ f2, const float* __restrict__ f3,
                         float* __restrict__ PB) {
  int g = blockIdx.x * blockDim.x + threadIdx.x;     // 1024*64 threads
  int pair = g >> 6;          // i2*16 + i3
  int e    = g & 63;          // a*8 + b
  int i2 = pair >> 4, i3 = pair & 15;
  int a = e >> 3, b = e & 7;
  float s = 0.f;
#pragma unroll
  for (int c = 0; c < RANK; ++c)
    s += f2[b * (N2 * RANK) + i2 * RANK + c] * f3[c * (N3 * RANK) + i3 * RANK + a];
  PB[(size_t)pair * 64 + e] = s;
}

// ---------------------------------------------------------------------------
// Stage 1c: build W TRANSPOSED in f16:  Wt[col][row] = W[row][col]
//   W[row][col] = dot64(PA[(i0,i1)], PB[(i2,i3)])
//   row = i0*32 + i1/2 ; col = (i1&1)*1024 + i2*16 + i3
// Thread mapping has `row` fast-varying so the Wt store is coalesced.
// ---------------------------------------------------------------------------
__global__ void tr_build_wt(const float* __restrict__ PA, const float* __restrict__ PB,
                            _Float16* __restrict__ Wt) {
  int g = blockIdx.x * blockDim.x + threadIdx.x;     // 2048*2048 threads
  int row = g & (KDIM - 1);
  int col = g >> 11;
  int i0 = row >> 5;
  int i1 = ((row & 31) << 1) | (col >> 10);
  int i2 = (col >> 4) & 63;
  int i3 = col & 15;
  const float* pa = PA + (size_t)(i0 * 64 + i1) * 64;
  const float* pb = PB + (size_t)(i2 * 16 + i3) * 64;
  float s = 0.f;
#pragma unroll 8
  for (int e = 0; e < 64; ++e) s += pa[e] * pb[e];
  Wt[(size_t)col * KDIM + row] = (_Float16)s;
}

// ---------------------------------------------------------------------------
// Stage 1d: x (fp32) -> xh (f16), vectorized x4
// ---------------------------------------------------------------------------
__global__ void tr_conv_x(const float* __restrict__ x, _Float16* __restrict__ xh) {
  int g = blockIdx.x * blockDim.x + threadIdx.x;     // (8192*2048)/4 threads
  f4 v = *(const f4*)(x + (size_t)g * 4);
  _Float16* o = xh + (size_t)g * 4;
  o[0] = (_Float16)v.x; o[1] = (_Float16)v.y;
  o[2] = (_Float16)v.z; o[3] = (_Float16)v.w;
}

// ---------------------------------------------------------------------------
// Stage 2: C(8192x2048,f32) = xh(8192x2048,f16) * Wt^T(2048x2048,f16)
// Wt is W transposed ([n][k]) so BOTH tiles stage with contiguous b128 ops.
// Block = 256 threads = 8 wave32 waves; block tile 128(M) x 128(N).
// Waves arranged 2(M) x 4(N); each wave: 64x32 = 4x2 tiles of 16x16.
// K staged 64 deep -> 2 WMMA K-steps (16 v_wmma per wave) per barrier pair.
// ---------------------------------------------------------------------------
__global__ __launch_bounds__(256) void tr_gemm(
    const _Float16* __restrict__ xh, const _Float16* __restrict__ Wt,
    float* __restrict__ out) {
  __shared__ _Float16 As[128 * KTILE];  // x tile, [m][k]
  __shared__ _Float16 Bs[128 * KTILE];  // W tile, [n][k] (already transposed)

  const int t    = threadIdx.x;
  const int lane = t & 31;
  const int w    = t >> 5;
  const int wm   = w & 1;             // 0..1
  const int wn   = w >> 1;            // 0..3
  const int mBase = blockIdx.y * 128;
  const int nBase = blockIdx.x * 128;

  const int lm   = lane & 15;         // row/col within 16x16 tile
  const int khA  = (lane >> 4) * 8;   // A-frag K sub-offset per layout
  const int khB  = (lane >> 4) * 8;   // B-frag K sub-offset per layout

  v8f acc[4][2];
#pragma unroll
  for (int i = 0; i < 4; ++i)
#pragma unroll
    for (int j = 0; j < 2; ++j) acc[i][j] = (v8f){};

  for (int kk = 0; kk < KDIM; kk += KTILE) {
    // ---- stage both tiles: 1024 chunks of 8 f16 (16B) each, 4 per thread ----
#pragma unroll
    for (int i = 0; i < 4; ++i) {
      int c  = t + i * 256;
      int r  = c >> 3;                // 0..127
      int k8 = (c & 7) * 8;           // 0..56
      *(h8*)&As[r * KTILE + k8] =
          *(const h8*)(xh + (size_t)(mBase + r) * KDIM + kk + k8);
      *(h8*)&Bs[r * KTILE + k8] =
          *(const h8*)(Wt + (size_t)(nBase + r) * KDIM + kk + k8);
    }
    __syncthreads();

#pragma unroll
    for (int ks = 0; ks < KTILE; ks += 32) {
      // ---- build fragments (documented 16-bit A/B VGPR striping) ----
      v16h afr[4];
#pragma unroll
      for (int mt = 0; mt < 4; ++mt) {
        int row = wm * 64 + mt * 16 + lm;
        h8 lo = *(const h8*)&As[row * KTILE + ks + khA];       // K = khA..+7
        h8 hi = *(const h8*)&As[row * KTILE + ks + khA + 16];  // K = khA+16..+23
        v16h a;
#pragma unroll
        for (int j = 0; j < 8; ++j) { a[j] = lo[j]; a[8 + j] = hi[j]; }
        afr[mt] = a;
      }
      v16h bfr[2];
#pragma unroll
      for (int nt = 0; nt < 2; ++nt) {
        int col = wn * 32 + nt * 16 + lm;
        h8 lo = *(const h8*)&Bs[col * KTILE + ks + khB];       // K = khB..+7
        h8 hi = *(const h8*)&Bs[col * KTILE + ks + khB + 16];  // K = khB+16..+23
        v16h b;
#pragma unroll
        for (int j = 0; j < 8; ++j) { b[j] = lo[j]; b[8 + j] = hi[j]; }
        bfr[nt] = b;
      }

      // ---- 8 WMMAs per K-step, 16 per barrier ----
#pragma unroll
      for (int mt = 0; mt < 4; ++mt)
#pragma unroll
        for (int nt = 0; nt < 2; ++nt)
          acc[mt][nt] = __builtin_amdgcn_wmma_f32_16x16x32_f16(
              false, afr[mt], false, bfr[nt], (short)0, acc[mt][nt], false, false);
    }
    __syncthreads();
  }

  // ---- epilogue: C/D layout — VGPR r: lanes 0-15 M=r, lanes 16-31 M=r+8 ----
  const int ncol = nBase + wn * 32 + lm;
  const int mrow0 = mBase + wm * 64 + (lane >> 4) * 8;
#pragma unroll
  for (int mt = 0; mt < 4; ++mt)
#pragma unroll
    for (int nt = 0; nt < 2; ++nt)
#pragma unroll
      for (int r = 0; r < 8; ++r)
        out[(size_t)(mrow0 + mt * 16 + r) * NDIM + (ncol + nt * 16)] =
            acc[mt][nt][r];
}

// ---------------------------------------------------------------------------
extern "C" void kernel_launch(void* const* d_in, const int* in_sizes, int n_in,
                              void* d_out, int out_size, void* d_ws, size_t ws_size,
                              hipStream_t stream) {
  const float* x  = (const float*)d_in[0];
  const float* f0 = (const float*)d_in[1];
  const float* f1 = (const float*)d_in[2];
  const float* f2 = (const float*)d_in[3];
  const float* f3 = (const float*)d_in[4];
  float* out = (float*)d_out;

  // workspace layout (256B aligned)
  char* ws = (char*)d_ws;
  float*    PA = (float*)(ws);                               // 4096*64*4 = 1 MB
  float*    PB = (float*)(ws + (1u << 20));                  // 1024*64*4 = 256 KB
  _Float16* Wt = (_Float16*)(ws + (1u << 20) + (256u << 10));// 2048*2048*2 = 8 MB
  _Float16* xh = (_Float16*)(ws + (1u << 20) + (256u << 10) + (8u << 20)); // 32 MB

  tr_pairA   <<<(4096 * 64) / 256, 256, 0, stream>>>(f0, f1, PA);
  tr_pairB   <<<(1024 * 64) / 256, 256, 0, stream>>>(f2, f3, PB);
  tr_build_wt<<<(NDIM * KDIM) / 256, 256, 0, stream>>>(PA, PB, Wt);
  tr_conv_x  <<<((size_t)MDIM * KDIM / 4) / 256, 256, 0, stream>>>(x, xh);

  dim3 grid(NDIM / 128, MDIM / 128);  // (16, 64)
  tr_gemm<<<grid, 256, 0, stream>>>(xh, Wt, out);
}
